// SharedKeysSplitModel_35588099014958
// MI455X (gfx1250) — matrix operand order repaired and checked by reading, not compiled, not run on hardware
//
#include <hip/hip_runtime.h>
#include <hip/hip_bf16.h>

// ---------------- constants ----------------
#define Bsz   64
#define Lsz   2048
#define Vsz   1024
#define Hsz   256
#define H2sz  512
#define HALFs 128

// ---------------- WMMA types ----------------
typedef __attribute__((ext_vector_type(16))) __bf16 v16bf;
typedef __attribute__((ext_vector_type(8)))  __bf16 v8bf;
typedef __attribute__((ext_vector_type(8)))  float  v8f;

union U16BF { v16bf v; v8bf h[2]; };

static __device__ inline v8f wmma_bf16(v16bf a, v16bf b, v8f c) {
  // (neg_a, A, neg_b, B, c_mod, C, reuse_a, reuse_b)
  return __builtin_amdgcn_wmma_f32_16x16x32_bf16(false, a, false, b, (short)0, c, false, false);
}

// A-fragment (16x32, row-major source with row stride `ld` halves).
// lane<16: row=lane, K = k0 + {0..7, 16..23}
// lane>=16: row=lane-16, K = k0 + {8..15, 24..31}
static __device__ inline v16bf loadA(const __bf16* base, int ld, int k0, int lane) {
  const int row = lane & 15;
  const int kb  = (lane >> 4) << 3;  // 0 or 8
  const __bf16* p = base + row * ld + k0 + kb;
  U16BF u;
  u.h[0] = *(const v8bf*)(p);
  u.h[1] = *(const v8bf*)(p + 16);
  return u.v;
}

// B-fragment (32x16) from an N-major (transposed) weight: wT[n][k], row stride ldk halves.
// lane holds column n0+(lane&15), K = k0 + (lane/16)*8 + {0..7, 16..23}
static __device__ inline v16bf loadB(const __bf16* wT, int ldk, int n0, int k0, int lane) {
  const __bf16* p = wT + (size_t)(n0 + (lane & 15)) * ldk + k0 + ((lane >> 4) << 3);
  U16BF u;
  u.h[0] = *(const v8bf*)(p);
  u.h[1] = *(const v8bf*)(p + 16);
  return u.v;
}

// Async copy: 512 bytes (32 lanes x 16B) global -> LDS, tracked by ASYNCcnt.
static __device__ inline void async_copy_b128(unsigned lds_addr, const void* gaddr) {
  asm volatile("global_load_async_to_lds_b128 %0, %1, off"
               :: "v"(lds_addr), "v"((unsigned long long)(uintptr_t)gaddr)
               : "memory");
}
static __device__ inline void wait_asynccnt_le1() {
  asm volatile("s_wait_asynccnt 0x1" ::: "memory");
}
static __device__ inline void wait_asynccnt_0() {
  asm volatile("s_wait_asynccnt 0x0" ::: "memory");
}

// ---------------- prep: weight transpose + bf16 convert ----------------
// dst[n*K + k] = (bf16) src[k*N + n]
__global__ void k_transpose(const float* __restrict__ src, __bf16* __restrict__ dst, int K, int N) {
  int idx = blockIdx.x * blockDim.x + threadIdx.x;
  if (idx >= K * N) return;
  int k = idx / N;
  int n = idx - k * N;
  dst[(size_t)n * K + k] = (__bf16)src[idx];
}

// ---------------- kernel 1: fused token pipeline ----------------
// One wave per 16-token tile: gather embed -> GEMM1(relu) -> GEMM2(+residual)
// -> LayerNorm -> GEMM3 -> proj (f32, token-major [B*L][H]).
#define WAVE_LDS 49280   // e_bf 8192 | g1_bf 16384 | x_f 16384 | h_bf 8192 | sid 128

__global__ __launch_bounds__(128) void k_tokens(
    const int*   __restrict__ seq,   const float* __restrict__ embed,
    const __bf16* __restrict__ w1T,  const float* __restrict__ b1,
    const __bf16* __restrict__ w2T,  const float* __restrict__ b2,
    const float* __restrict__ ln_g,  const float* __restrict__ ln_b,
    const __bf16* __restrict__ kpT,  const float* __restrict__ kp_b,
    float* __restrict__ proj)
{
  extern __shared__ char smem[];
  const int wave = threadIdx.x >> 5;
  const int lane = threadIdx.x & 31;
  char* my = smem + (size_t)wave * WAVE_LDS;
  __bf16* e_bf  = (__bf16*)(my);
  __bf16* g1_bf = (__bf16*)(my + 8192);
  float*  x_f   = (float*) (my + 24576);
  __bf16* h_bf  = (__bf16*)(my + 40960);
  int*    sid   = (int*)   (my + 49152);

  const int tile0 = (blockIdx.x * 4 + wave) * 16;
  const int row = lane & 15;
  const int c0  = (lane >> 4) * 128;   // each lane covers half the columns of its row
  const int r0  = (lane >> 4) << 3;    // C-fragment row base

  // ---- stage gathered embeddings (bf16) + token ids ----
  {
    const int tok = tile0 + row;
    const int id  = seq[tok];
    if (lane < 16) sid[row] = id;
    const float* ep = embed + (size_t)id * Hsz + c0;
    #pragma unroll 8
    for (int c = 0; c < 128; c += 4) {
      float4 f = *(const float4*)(ep + c);
      __bf16* d = e_bf + row * Hsz + c0 + c;
      d[0] = (__bf16)f.x; d[1] = (__bf16)f.y; d[2] = (__bf16)f.z; d[3] = (__bf16)f.w;
    }
  }
  __syncthreads();

  // ---- GEMM1: g1 = relu(e @ w1 + b1)   (16 x 512) ----
  for (int nt = 0; nt < 32; ++nt) {
    const int n0 = nt * 16;
    v8f acc = {0.f,0.f,0.f,0.f,0.f,0.f,0.f,0.f};
    #pragma unroll
    for (int kb = 0; kb < 8; ++kb) {
      v16bf a = loadA(e_bf, Hsz, kb * 32, lane);
      v16bf b = loadB(w1T, Hsz, n0, kb * 32, lane);
      acc = wmma_bf16(a, b, acc);
    }
    const int col = n0 + (lane & 15);
    const float bias = b1[col];
    #pragma unroll
    for (int r = 0; r < 8; ++r) {
      float v = acc[r] + bias;
      v = v > 0.f ? v : 0.f;
      g1_bf[(r0 + r) * H2sz + col] = (__bf16)v;
    }
  }
  __syncthreads();

  // ---- GEMM2: x = g1 @ w2 + b2 + e   (16 x 256, f32) ----
  for (int nt = 0; nt < 16; ++nt) {
    const int n0 = nt * 16;
    v8f acc = {0.f,0.f,0.f,0.f,0.f,0.f,0.f,0.f};
    #pragma unroll
    for (int kb = 0; kb < 16; ++kb) {
      v16bf a = loadA(g1_bf, H2sz, kb * 32, lane);
      v16bf b = loadB(w2T, H2sz, n0, kb * 32, lane);
      acc = wmma_bf16(a, b, acc);
    }
    const int col = n0 + (lane & 15);
    const float bias = b2[col];
    #pragma unroll
    for (int r = 0; r < 8; ++r) {
      const int rr = r0 + r;
      const float e_res = embed[(size_t)sid[rr] * Hsz + col];
      x_f[rr * Hsz + col] = acc[r] + bias + e_res;
    }
  }
  __syncthreads();

  // ---- LayerNorm: h = (x - mu) * rsqrt(var + 1e-5) * g + b  (bf16) ----
  {
    float s = 0.f, s2 = 0.f;
    const float* xr = x_f + row * Hsz + c0;
    #pragma unroll 8
    for (int c = 0; c < 128; ++c) { float v = xr[c]; s += v; s2 += v * v; }
    s  += __shfl_xor(s, 16);
    s2 += __shfl_xor(s2, 16);
    const float mu  = s * (1.f / 256.f);
    const float var = s2 * (1.f / 256.f) - mu * mu;
    const float rs  = rsqrtf(var + 1e-5f);
    #pragma unroll 8
    for (int c = 0; c < 128; ++c) {
      const int cc = c0 + c;
      const float v = (xr[c] - mu) * rs * ln_g[cc] + ln_b[cc];
      h_bf[row * Hsz + cc] = (__bf16)v;
    }
  }
  __syncthreads();

  // ---- GEMM3: proj = h @ kp_w + kp_b ----
  for (int nt = 0; nt < 16; ++nt) {
    const int n0 = nt * 16;
    v8f acc = {0.f,0.f,0.f,0.f,0.f,0.f,0.f,0.f};
    #pragma unroll
    for (int kb = 0; kb < 8; ++kb) {
      v16bf a = loadA(h_bf, Hsz, kb * 32, lane);
      v16bf b = loadB(kpT, Hsz, n0, kb * 32, lane);
      acc = wmma_bf16(a, b, acc);
    }
    const int col = n0 + (lane & 15);
    const float bias = kp_b[col];
    #pragma unroll
    for (int r = 0; r < 8; ++r)
      proj[(size_t)(tile0 + r0 + r) * Hsz + col] = acc[r] + bias;
  }
}

// ---------------- kernel 2: sequential delta-rule scan ----------------
// One WG per (batch, half). 256 threads; thread t owns M[row = t&127][jh..jh+63]
// in 64 VGPRs (jh = (t>>7)*64). k broadcast via LDS.
// k-vectors are double-buffered in LDS and prefetched one step ahead by wave 0
// using GLOBAL_LOAD_ASYNC_TO_LDS_B128 (ASYNCcnt), removing the global-load
// latency from the 2047-step serial dependency chain.
__global__ __launch_bounds__(256) void k_scan(const float* __restrict__ proj,
                                              float* __restrict__ ctx)
{
  const int b    = blockIdx.x >> 1;
  const int half = blockIdx.x & 1;
  __shared__ float kvbuf[2][HALFs];
  __shared__ float vpp[256];
  __shared__ float dvs[HALFs];
  __shared__ float nsh[2];
  const int t   = threadIdx.x;
  const int row = t & 127;
  const int jh  = (t >> 7) << 6;   // 0 or 64

  float M[64];
  #pragma unroll
  for (int j = 0; j < 64; ++j) M[j] = 0.f;

  const float* base = proj + (size_t)b * Lsz * Hsz + half * HALFs;
  const float inv_l = 1.0f / (float)Lsz;

  // prologue: async-prefetch k[0] into buffer 0 (wave 0; 32 lanes x 16B = 512B)
  const unsigned lds0 = (unsigned)(uintptr_t)(&kvbuf[0][0]) + (unsigned)(t * 16);
  const unsigned lds1 = (unsigned)(uintptr_t)(&kvbuf[1][0]) + (unsigned)(t * 16);
  if (t < 32) async_copy_b128(lds0, base + t * 4);

  for (int step = 0; step < Lsz - 1; ++step) {
    const int cur = step & 1;
    if (t < 32) {
      // prefetch next k-vector (step+1; step 2047 is q) into the other buffer
      async_copy_b128(cur ? lds0 : lds1, base + (size_t)(step + 1) * Hsz + t * 4);
      // async loads complete in order: <=1 outstanding means current buf is ready
      wait_asynccnt_le1();
    }
    __syncthreads();
    const float* kv = kvbuf[cur];
    float acc = 0.f, sq = 0.f;
    #pragma unroll 16
    for (int j = 0; j < 64; ++j) { const float k = kv[jh + j]; acc += M[j] * k; sq += k * k; }
    vpp[t] = acc;
    if (t == 0)   nsh[0] = sq;
    if (t == 128) nsh[1] = sq;
    __syncthreads();
    if (t < 128) {
      const float vp   = vpp[row] + vpp[128 + row];
      const float norm = nsh[0] + nsh[1] + 1e-6f;
      const float dv   = kv[row] - vp / norm;
      const float sc   = half ? (float)(step + 1) * inv_l : 1.0f;
      dvs[row] = dv * sc;
    }
    __syncthreads();
    const float d = dvs[row];
    #pragma unroll 16
    for (int j = 0; j < 64; ++j) M[j] += d * kv[jh + j];
    __syncthreads();   // readers done before next iteration's prefetch reuses buffer
  }

  // context: ctx = M @ q ; q (row 2047) was prefetched into buffer 1
  if (t < 32) wait_asynccnt_0();
  __syncthreads();
  const float* kv = kvbuf[1];
  float acc = 0.f;
  #pragma unroll 16
  for (int j = 0; j < 64; ++j) acc += M[j] * kv[jh + j];
  vpp[t] = acc;
  __syncthreads();
  if (t < 128) ctx[(size_t)b * Hsz + half * HALFs + row] = vpp[row] + vpp[128 + row];
}

// ---------------- kernel 3: out = ctx @ out_w + out_b (WMMA) ----------------
__global__ __launch_bounds__(256) void k_out(const float* __restrict__ ctx,
                                             const __bf16* __restrict__ outT,
                                             const float* __restrict__ out_b,
                                             float* __restrict__ out)
{
  const int wave = threadIdx.x >> 5;
  const int lane = threadIdx.x & 31;
  const int tile = blockIdx.x * 8 + wave;       // 0..255
  const int m0 = (tile >> 6) * 16;              // 4 row tiles
  const int n0 = (tile & 63) * 16;              // 64 col tiles

  v8f acc = {0.f,0.f,0.f,0.f,0.f,0.f,0.f,0.f};
  const int row = lane & 15;
  const int kb8 = (lane >> 4) << 3;
  #pragma unroll
  for (int kb = 0; kb < 8; ++kb) {
    const int k0 = kb * 32;
    // A fragment: convert ctx f32 rows on the fly
    const float* ap = ctx + (size_t)(m0 + row) * Hsz + k0 + kb8;
    U16BF a;
    #pragma unroll
    for (int i = 0; i < 8; ++i) { a.v[i] = (__bf16)ap[i]; a.v[i + 8] = (__bf16)ap[16 + i]; }
    v16bf b = loadB(outT, Hsz, n0, k0, lane);
    acc = wmma_bf16(a.v, b, acc);
  }
  const int col = n0 + (lane & 15);
  const float bias = out_b[col];
  const int r0 = (lane >> 4) << 3;
  #pragma unroll
  for (int r = 0; r < 8; ++r)
    out[(size_t)(m0 + r0 + r) * Vsz + col] = acc[r] + bias;
}

// ---------------- host launcher ----------------
extern "C" void kernel_launch(void* const* d_in, const int* in_sizes, int n_in,
                              void* d_out, int out_size, void* d_ws, size_t ws_size,
                              hipStream_t stream) {
  const int*   seq   = (const int*)  d_in[0];
  const float* embed = (const float*)d_in[1];
  const float* w1    = (const float*)d_in[2];
  const float* b1    = (const float*)d_in[3];
  const float* w2    = (const float*)d_in[4];
  const float* b2    = (const float*)d_in[5];
  const float* ln_g  = (const float*)d_in[6];
  const float* ln_b  = (const float*)d_in[7];
  const float* kp_w  = (const float*)d_in[8];
  const float* kp_b  = (const float*)d_in[9];
  const float* out_w = (const float*)d_in[10];
  const float* out_b = (const float*)d_in[11];

  char* ws = (char*)d_ws;
  const size_t projBytes = (size_t)Bsz * Lsz * Hsz * 4;           // 134,217,728
  float*  proj = (float*)ws;
  __bf16* w1T  = (__bf16*)(ws + projBytes);                        // 512x256
  __bf16* w2T  = (__bf16*)(ws + projBytes + 262144);               // 256x512
  __bf16* kpT  = (__bf16*)(ws + projBytes + 524288);               // 256x256
  __bf16* outT = (__bf16*)(ws + projBytes + 655360);               // 1024x256
  float*  ctx  = (float*)(ws + projBytes + 1179648);               // 64x256

  // prep: bf16 transposed weights
  k_transpose<<<(Hsz * H2sz + 255) / 256, 256, 0, stream>>>(w1, w1T, Hsz, H2sz);
  k_transpose<<<(H2sz * Hsz + 255) / 256, 256, 0, stream>>>(w2, w2T, H2sz, Hsz);
  k_transpose<<<(Hsz * Hsz + 255) / 256, 256, 0, stream>>>(kp_w, kpT, Hsz, Hsz);
  k_transpose<<<(Hsz * Vsz + 255) / 256, 256, 0, stream>>>(out_w, outT, Hsz, Vsz);

  // fused token pipeline: 131072 tokens / (4 waves * 16 tokens) = 2048 blocks
  k_tokens<<<2048, 128, 4 * WAVE_LDS, stream>>>(seq, embed, w1T, b1, w2T, b2,
                                                ln_g, ln_b, kpT, kp_b, proj);

  // sequential scan: one WG per (batch, half)
  k_scan<<<Bsz * 2, 256, 0, stream>>>(proj, ctx);

  // output GEMM
  k_out<<<32, 256, 0, stream>>>(ctx, outT, out_b, (float*)d_out);
}
